// SelfAttention_6682969113297
// MI455X (gfx1250) — matrix operand order repaired and checked by reading, not compiled, not run on hardware
//
#include <hip/hip_runtime.h>
#include <math.h>

// ---------------------------------------------------------------------------
// Transformer block forward for MI455X (gfx1250, wave32, WMMA).
// bf16 operands + fp32 accumulation via v_wmma_f32_16x16x32_bf16.
// - Weights pre-transposed to [N][K] so B fragments are contiguous loads.
// - GEMM and attention stage block-shared tiles into LDS with double
//   buffering via GLOBAL_LOAD_ASYNC_TO_LDS_B128 (ASYNCcnt-tracked).
// - Attention is flash-style; softmax row-sums accumulated via an extra
//   WMMA against an all-ones fragment (keeps stats in D-tile layout).
// ---------------------------------------------------------------------------

typedef __bf16 bf16;
typedef __attribute__((ext_vector_type(16))) __bf16 v16bf;
typedef __attribute__((ext_vector_type(8)))  __bf16 bf16x8;
typedef __attribute__((ext_vector_type(8)))  float  v8f;
typedef __attribute__((ext_vector_type(4)))  int    v4i;

#define DIM   512
#define SEQ   2048
#define BATCH 4
#define NTOK  (BATCH * SEQ)
#define NHEAD 8
#define HDIM  64
#define FFN_D 2048

// ---- async global->LDS copy (CDNA5), with portable fallback ---------------
typedef __attribute__((address_space(1))) v4i gv4i_t;
typedef __attribute__((address_space(3))) v4i lv4i_t;

#if __has_builtin(__builtin_amdgcn_global_load_async_to_lds_b128)
#define HAS_ASYNC_LDS 1
#else
#define HAS_ASYNC_LDS 0
#endif

__device__ inline void async_copy_b128(const bf16* __restrict__ gsrc, bf16* ldst) {
#if HAS_ASYNC_LDS
  __builtin_amdgcn_global_load_async_to_lds_b128((gv4i_t*)gsrc, (lv4i_t*)ldst, 0, 0);
#else
  *reinterpret_cast<bf16x8*>(ldst) = *reinterpret_cast<const bf16x8*>(gsrc);
#endif
}

__device__ inline void async_wait() {
#if HAS_ASYNC_LDS
#if __has_builtin(__builtin_amdgcn_s_wait_asynccnt)
  __builtin_amdgcn_s_wait_asynccnt(0);
#else
  asm volatile("s_wait_asynccnt 0x0" ::: "memory");
#endif
#endif
}

// ---- WMMA fragment loaders (layouts per CDNA5 ISA 7.12.2, wave32) ---------
// A fragment 16x32 bf16: lane L holds row (L&15); lanes 0-15 carry K 0-7 and
// 16-23, lanes 16-31 carry K 8-15 and 24-31 -> two contiguous 8-elem chunks.
__device__ inline v16bf load_a_frag(const bf16* __restrict__ base, int ld, int k0) {
  int lane = threadIdx.x & 31;
  int row  = lane & 15;
  int koff = (lane < 16) ? 0 : 8;
  const bf16* p = base + (size_t)row * ld + k0;
  union { v16bf v; bf16x8 h[2]; } u;
  u.h[0] = *reinterpret_cast<const bf16x8*>(p + koff);
  u.h[1] = *reinterpret_cast<const bf16x8*>(p + 16 + koff);
  return u.v;
}

// B fragment 32x16 bf16: lane L holds column (L&15); lanes 0-15 carry K 0-7
// and 16-23, lanes 16-31 carry K 8-15 and 24-31 (matches A fragment) when the
// matrix is stored column-major (stride = colstride along K).
__device__ inline v16bf load_b_frag(const bf16* __restrict__ colbase, size_t colstride, int k0) {
  int lane = threadIdx.x & 31;
  const bf16* p = colbase + (size_t)(lane & 15) * colstride + k0 + ((lane < 16) ? 0 : 8);
  union { v16bf v; bf16x8 h[2]; } u;
  u.h[0] = *reinterpret_cast<const bf16x8*>(p);
  u.h[1] = *reinterpret_cast<const bf16x8*>(p + 16);
  return u.v;
}

// ---- conversion / transpose kernels ---------------------------------------
__global__ void cvt_bf16_kernel(const float* __restrict__ src, bf16* __restrict__ dst, int n) {
  int i = blockIdx.x * 256 + threadIdx.x;
  if (i < n) dst[i] = (bf16)src[i];
}

// src[K][N] fp32 -> dst[N][K] bf16  (weights are small; perf irrelevant)
__global__ void transpose_cvt_kernel(const float* __restrict__ src, bf16* __restrict__ dst,
                                     int K, int N) {
  int i = blockIdx.x * 256 + threadIdx.x;
  if (i < K * N) {
    int k = i / N, n = i % N;
    dst[(size_t)n * K + k] = (bf16)src[i];
  }
}

// ---- generic WMMA GEMM: C[M,N] = A[M,K](bf16) * Bt[N,K]^T + bias ----------
// 8 waves / block; wave w -> rows [blkX*128 + w*16), cols [blkY*64, +64).
// B tile (64 cols x 32 k = 4KB) is shared by all waves -> staged in LDS with
// async copies, double-buffered across the K loop.
template<int OUT_BF16, int GELU, int STORE_VT>
__global__ __launch_bounds__(256) void gemm_kernel(
    const bf16* __restrict__ A, const bf16* __restrict__ Bt,
    const float* __restrict__ bias,
    float* __restrict__ Cf, bf16* __restrict__ Cb,
    int M, int N, int K)
{
  __shared__ alignas(16) bf16 bsh[2][64 * 32];

  int wave = threadIdx.x >> 5;
  int lane = threadIdx.x & 31;
  int m0 = blockIdx.x * 128 + wave * 16;
  int n0 = blockIdx.y * 64;
  (void)M;

  // each of 256 threads copies one 16B chunk of the 4KB B tile
  const int scol   = threadIdx.x >> 2;        // 0..63 (local column)
  const int schunk = (threadIdx.x & 3) * 8;   // 0,8,16,24 (k offset)

  auto stage = [&](int k0, int buf) {
    const bf16* src = Bt + (size_t)(n0 + scol) * K + k0 + schunk;
    async_copy_b128(src, &bsh[buf][scol * 32 + schunk]);
  };

  stage(0, 0);
  v8f acc[4] = {};
  int buf = 0;
  for (int k0 = 0; k0 < K; k0 += 32) {
    async_wait();
    __syncthreads();                          // bsh[buf] ready for all waves
    if (k0 + 32 < K) {
      stage(k0 + 32, buf ^ 1);                // fill other buffer
      __builtin_prefetch(A + (size_t)(m0 + (lane & 15)) * K + k0 + 32, 0, 3);
    }
    v16bf a = load_a_frag(A + (size_t)m0 * K, K, k0);
#pragma unroll
    for (int j = 0; j < 4; ++j) {
      v16bf b = load_b_frag(&bsh[buf][(j * 16) * 32], 32, 0);
      acc[j] = __builtin_amdgcn_wmma_f32_16x16x32_bf16(
          false, a, false, b, (short)0, acc[j], false, false);
    }
    buf ^= 1;
  }

  int col   = lane & 15;
  int rbase = (lane < 16) ? 0 : 8;   // D tile: vgpr i -> row i (lanes<16) / 8+i
#pragma unroll
  for (int j = 0; j < 4; ++j) {
    int n = n0 + j * 16 + col;
    float bv = bias[n];
#pragma unroll
    for (int i = 0; i < 8; ++i) {
      int m = m0 + rbase + i;
      float v = acc[j][i] + bv;
      if (GELU) v = 0.5f * v * (1.0f + erff(v * 0.70710678f));
      if (STORE_VT) {
        // store V transposed per batch: Vt[(b*DIM + n)*SEQ + tok]
        int bidx = m >> 11;          // / SEQ
        int tok  = m & (SEQ - 1);
        Cb[((size_t)bidx * DIM + n) * SEQ + tok] = (bf16)v;
      } else if (OUT_BF16) {
        Cb[(size_t)m * N + n] = (bf16)v;
      } else {
        Cf[(size_t)m * N + n] = v;
      }
    }
  }
}

// ---- fused flash attention -------------------------------------------------
// grid = (BATCH*NHEAD, SEQ/128); 8 waves/block; wave -> one 16-query block.
// All waves share (b,h) and march key blocks in lockstep, so K and V tiles
// (4KB each per 32-key step) are staged once per block in LDS via async
// copies, double-buffered — 8x less L2 traffic than per-wave loads.
__global__ __launch_bounds__(256) void attn_kernel(
    const bf16* __restrict__ Q, const bf16* __restrict__ Kb,
    const bf16* __restrict__ Vt, bf16* __restrict__ O)
{
  __shared__ alignas(16) bf16 ksh[2][32 * 64];    // [key_local][dim]   4KB x2
  __shared__ alignas(16) bf16 vsh[2][64 * 32];    // [dim_local][key]   4KB x2
  __shared__ alignas(16) bf16 pbuf[8][16 * 32];   // per-wave P tile (16x32)

  int wave = threadIdx.x >> 5;
  int lane = threadIdx.x & 31;
  int bh = blockIdx.x;
  int b  = bh >> 3, h = bh & 7;
  int q0 = (blockIdx.y * 8 + wave) * 16;
  const float scale = 0.125f;                     // 1/sqrt(64)
  int col = lane & 15;

  const bf16* qbase = Q + ((size_t)(b * SEQ + q0)) * DIM + h * HDIM;
  v16bf aq0 = load_a_frag(qbase, DIM, 0);         // head dims 0..31
  v16bf aq1 = load_a_frag(qbase, DIM, 32);        // head dims 32..63

  // all-ones B fragment: P @ ones accumulates row-sums in D-tile layout
  v16bf ones;
#pragma unroll
  for (int i = 0; i < 16; ++i) ones[i] = (bf16)1.0f;

  // staging assignments: K tile = 32 rows x 128B (8 chunks of 16B per row);
  // V tile = 64 rows x 64B (4 chunks of 16B per row). 256 threads each.
  const int krow = threadIdx.x >> 3, kchunk = (threadIdx.x & 7) * 8;
  const int vrow = threadIdx.x >> 2, vchunk = (threadIdx.x & 3) * 8;
  const bf16* kg = Kb + ((size_t)b * SEQ) * DIM + h * HDIM;        // [key][dim]
  const bf16* vg = Vt + ((size_t)(b * DIM + h * HDIM)) * SEQ;      // [dim][key]

  auto stage_kv = [&](int kb, int buf) {
    async_copy_b128(kg + (size_t)(kb + krow) * DIM + kchunk, &ksh[buf][krow * 64 + kchunk]);
    async_copy_b128(vg + (size_t)vrow * SEQ + kb + vchunk,   &vsh[buf][vrow * 32 + vchunk]);
  };

  v8f acc[4] = {};
  v8f lacc   = {};                                // softmax denominators
  float mrow[8];
#pragma unroll
  for (int i = 0; i < 8; ++i) mrow[i] = -1e30f;

  stage_kv(0, 0);
  int buf = 0;
  for (int kb = 0; kb < SEQ; kb += 32) {
    async_wait();
    __syncthreads();                              // ksh/vsh[buf] ready
    if (kb + 32 < SEQ) stage_kv(kb + 32, buf ^ 1);

    // --- scores: S[16x32] as two 16x16 D tiles from LDS K tile -------------
    v8f s[2];
#pragma unroll
    for (int j = 0; j < 2; ++j) {
      v16bf b0 = load_b_frag(&ksh[buf][(j * 16) * 64], 64, 0);
      v16bf b1 = load_b_frag(&ksh[buf][(j * 16) * 64], 64, 32);
      v8f z = {};
      z = __builtin_amdgcn_wmma_f32_16x16x32_bf16(false, aq0, false, b0, (short)0, z, false, false);
      z = __builtin_amdgcn_wmma_f32_16x16x32_bf16(false, aq1, false, b1, (short)0, z, false, false);
      s[j] = z;
    }

    // --- online softmax: row max via cross-lane reduce, rescale, exp -------
#pragma unroll
    for (int i = 0; i < 8; ++i) {
      float s0 = s[0][i] * scale;
      float s1 = s[1][i] * scale;
      float v  = fmaxf(s0, s1);
#pragma unroll
      for (int off = 8; off >= 1; off >>= 1) v = fmaxf(v, __shfl_xor(v, off, 32));
      float mn    = fmaxf(mrow[i], v);
      float alpha = __expf(mrow[i] - mn);
      mrow[i] = mn;
      float p0 = __expf(s0 - mn);
      float p1 = __expf(s1 - mn);
      lacc[i] = lacc[i] * alpha;
#pragma unroll
      for (int j = 0; j < 4; ++j) acc[j][i] = acc[j][i] * alpha;
      // write P elements (D layout) to LDS for D->A re-layout
      int r = ((lane < 16) ? 0 : 8) + i;
      pbuf[wave][r * 32 + col]      = (bf16)p0;
      pbuf[wave][r * 32 + 16 + col] = (bf16)p1;
    }
    __builtin_amdgcn_wave_barrier();              // DS ops are in-order per wave

    v16bf pa = load_a_frag(&pbuf[wave][0], 32, 0);

    // --- denominators: lacc += P @ ones ------------------------------------
    lacc = __builtin_amdgcn_wmma_f32_16x16x32_bf16(
        false, pa, false, ones, (short)0, lacc, false, false);

    // --- out += P(16x32) @ V(32x64) from LDS V tile -------------------------
#pragma unroll
    for (int j = 0; j < 4; ++j) {
      v16bf bv = load_b_frag(&vsh[buf][(j * 16) * 32], 32, 0);
      acc[j] = __builtin_amdgcn_wmma_f32_16x16x32_bf16(
          false, pa, false, bv, (short)0, acc[j], false, false);
    }
    buf ^= 1;
  }

  // --- epilogue: normalize and store bf16 -----------------------------------
#pragma unroll
  for (int i = 0; i < 8; ++i) {
    int r = ((lane < 16) ? 0 : 8) + i;
    float inv = 1.0f / lacc[i];
#pragma unroll
    for (int j = 0; j < 4; ++j) {
      O[((size_t)(b * SEQ + q0 + r)) * DIM + h * HDIM + j * 16 + col] =
          (bf16)(acc[j][i] * inv);
    }
  }
}

// ---- fused residual-add + LayerNorm ---------------------------------------
template<int WRITE_BF16>
__global__ __launch_bounds__(256) void add_ln_kernel(
    const float* __restrict__ A, const float* __restrict__ R,
    const float* __restrict__ g, const float* __restrict__ be,
    float* __restrict__ outf, bf16* __restrict__ outb)
{
  int row = blockIdx.x;
  int tid = threadIdx.x;
  size_t base = (size_t)row * DIM;

  float x0 = A[base + tid]       + R[base + tid];
  float x1 = A[base + tid + 256] + R[base + tid + 256];
  float s  = x0 + x1;
  float ss = x0 * x0 + x1 * x1;
#pragma unroll
  for (int off = 16; off >= 1; off >>= 1) {
    s  += __shfl_xor(s,  off, 32);
    ss += __shfl_xor(ss, off, 32);
  }
  __shared__ float sbuf[8], ssbuf[8];
  int wave = tid >> 5, lane = tid & 31;
  if (lane == 0) { sbuf[wave] = s; ssbuf[wave] = ss; }
  __syncthreads();
  if (wave == 0) {
    float a  = (lane < 8) ? sbuf[lane]  : 0.f;
    float b2 = (lane < 8) ? ssbuf[lane] : 0.f;
#pragma unroll
    for (int off = 16; off >= 1; off >>= 1) {
      a  += __shfl_xor(a,  off, 32);
      b2 += __shfl_xor(b2, off, 32);
    }
    if (lane == 0) { sbuf[0] = a; ssbuf[0] = b2; }
  }
  __syncthreads();
  float mean = sbuf[0] * (1.0f / DIM);
  float var  = ssbuf[0] * (1.0f / DIM) - mean * mean;
  float rstd = rsqrtf(var + 1e-5f);
  float y0 = (x0 - mean) * rstd * g[tid]       + be[tid];
  float y1 = (x1 - mean) * rstd * g[tid + 256] + be[tid + 256];
  outf[base + tid]       = y0;
  outf[base + tid + 256] = y1;
  if (WRITE_BF16) {
    outb[base + tid]       = (bf16)y0;
    outb[base + tid + 256] = (bf16)y1;
  }
}

// ---------------------------------------------------------------------------
extern "C" void kernel_launch(void* const* d_in, const int* in_sizes, int n_in,
                              void* d_out, int out_size, void* d_ws, size_t ws_size,
                              hipStream_t stream) {
  const float* x  = (const float*)d_in[0];
  const float* wq = (const float*)d_in[1];  const float* bq = (const float*)d_in[2];
  const float* wk = (const float*)d_in[3];  const float* bk = (const float*)d_in[4];
  const float* wv = (const float*)d_in[5];  const float* bv = (const float*)d_in[6];
  const float* wo = (const float*)d_in[7];  const float* bo = (const float*)d_in[8];
  const float* w1 = (const float*)d_in[9];  const float* b1 = (const float*)d_in[10];
  const float* w2 = (const float*)d_in[11]; const float* b2 = (const float*)d_in[12];
  const float* g1 = (const float*)d_in[13]; const float* be1 = (const float*)d_in[14];
  const float* g2 = (const float*)d_in[15]; const float* be2 = (const float*)d_in[16];
  float* out = (float*)d_out;

  // workspace layout (256B aligned regions)
  char* base = (char*)d_ws;
  size_t off = 0;
  auto take = [&](size_t bytes) -> char* {
    char* p = base + off;
    off = (off + bytes + 255) & ~(size_t)255;
    return p;
  };
  bf16* xb  = (bf16*)take((size_t)NTOK * DIM * 2);
  bf16* wqt = (bf16*)take((size_t)DIM * DIM * 2);
  bf16* wkt = (bf16*)take((size_t)DIM * DIM * 2);
  bf16* wvt = (bf16*)take((size_t)DIM * DIM * 2);
  bf16* wot = (bf16*)take((size_t)DIM * DIM * 2);
  bf16* w1t = (bf16*)take((size_t)DIM * FFN_D * 2);   // [FFN][DIM]
  bf16* w2t = (bf16*)take((size_t)FFN_D * DIM * 2);   // [DIM][FFN]
  bf16* qb  = (bf16*)take((size_t)NTOK * DIM * 2);
  bf16* kbm = (bf16*)take((size_t)NTOK * DIM * 2);
  bf16* vt  = (bf16*)take((size_t)NTOK * DIM * 2);    // [b][DIM][SEQ]
  bf16* ob  = (bf16*)take((size_t)NTOK * DIM * 2);
  float* att = (float*)take((size_t)NTOK * DIM * 4);  // reused as ff output
  float* x1f = (float*)take((size_t)NTOK * DIM * 4);
  bf16*  x1b = (bf16*)take((size_t)NTOK * DIM * 2);
  bf16*  hb  = (bf16*)take((size_t)NTOK * FFN_D * 2);
  (void)ws_size; (void)in_sizes; (void)n_in; (void)out_size;

  // 1) precision conversion / weight transposes
  cvt_bf16_kernel<<<(NTOK * DIM) / 256, 256, 0, stream>>>(x, xb, NTOK * DIM);
  transpose_cvt_kernel<<<(DIM * DIM) / 256, 256, 0, stream>>>(wq, wqt, DIM, DIM);
  transpose_cvt_kernel<<<(DIM * DIM) / 256, 256, 0, stream>>>(wk, wkt, DIM, DIM);
  transpose_cvt_kernel<<<(DIM * DIM) / 256, 256, 0, stream>>>(wv, wvt, DIM, DIM);
  transpose_cvt_kernel<<<(DIM * DIM) / 256, 256, 0, stream>>>(wo, wot, DIM, DIM);
  transpose_cvt_kernel<<<(DIM * FFN_D) / 256, 256, 0, stream>>>(w1, w1t, DIM, FFN_D);
  transpose_cvt_kernel<<<(FFN_D * DIM) / 256, 256, 0, stream>>>(w2, w2t, FFN_D, DIM);

  // 2) QKV projections (V stored transposed for attention B-fragments)
  dim3 gqkv(NTOK / 128, DIM / 64);
  gemm_kernel<1, 0, 0><<<gqkv, 256, 0, stream>>>(xb, wqt, bq, nullptr, qb,  NTOK, DIM, DIM);
  gemm_kernel<1, 0, 0><<<gqkv, 256, 0, stream>>>(xb, wkt, bk, nullptr, kbm, NTOK, DIM, DIM);
  gemm_kernel<1, 0, 1><<<gqkv, 256, 0, stream>>>(xb, wvt, bv, nullptr, vt,  NTOK, DIM, DIM);

  // 3) fused flash attention
  attn_kernel<<<dim3(BATCH * NHEAD, SEQ / 128), 256, 0, stream>>>(qb, kbm, vt, ob);

  // 4) output projection (fp32 out)
  gemm_kernel<0, 0, 0><<<gqkv, 256, 0, stream>>>(ob, wot, bo, att, nullptr, NTOK, DIM, DIM);

  // 5) residual + LN1 (fp32 + bf16 out)
  add_ln_kernel<1><<<NTOK, 256, 0, stream>>>(att, x, g1, be1, x1f, x1b);

  // 6) FFN: GEMM+GELU, then GEMM (ff reuses att region)
  gemm_kernel<1, 1, 0><<<dim3(NTOK / 128, FFN_D / 64), 256, 0, stream>>>(
      x1b, w1t, b1, nullptr, hb, NTOK, FFN_D, DIM);
  gemm_kernel<0, 0, 0><<<gqkv, 256, 0, stream>>>(hb, w2t, b2, att, nullptr, NTOK, DIM, FFN_D);

  // 7) residual + LN2 -> final output
  add_ln_kernel<0><<<NTOK, 256, 0, stream>>>(att, x1f, g2, be2, out, nullptr);
}
